// LineSegDetector_38397007626869
// MI455X (gfx1250) — matrix operand order repaired and checked by reading, not compiled, not run on hardware
//
#include <hip/hip_runtime.h>
#include <hip/hip_bf16.h>
#include <math.h>

typedef __attribute__((ext_vector_type(16))) _Float16 v16h;
typedef __attribute__((ext_vector_type(8)))  _Float16 v8h;
typedef __attribute__((ext_vector_type(8)))  float    v8f;

#define CFEAT  64
#define HH     512
#define WW     512
#define HWSZ   (HH*WW)
#define NTOP   64
#define NLINES 2016      /* 64*63/2 */
#define NEGSL  0.2f
#define BNEPS  1e-5f

/* padded spatial geometries (1-px zero halo) */
#define X1P    10        /* conv1 input: 32x8 -> 34x10 */
#define Y1P    34
#define SP1    (Y1P*X1P) /* 340 */
#define X2P    6         /* conv2 input: 16x4 -> 18x6 */
#define Y2P    18
#define SP2    (Y2P*X2P) /* 108 */

#define WMMA16(A,B,C) __builtin_amdgcn_wmma_f32_16x16x32_f16(false, A, false, B, (short)0, C, false, false)

/* ---------------- stage 0: sigmoid + NMS + top-k + sampling ---------------- */

__global__ void k_sigmoid(const float* __restrict__ logits, float* __restrict__ sig) {
    int i = blockIdx.x * blockDim.x + threadIdx.x;
    if (i < HWSZ) sig[i] = 1.f / (1.f + expf(-logits[i]));
}

__global__ void k_nms(const float* __restrict__ sig, float* __restrict__ nms) {
    int i = blockIdx.x * blockDim.x + threadIdx.x;
    if (i >= HWSZ) return;
    int y = i / WW, x = i % WW;
    float s = sig[i], mx = -INFINITY;
    for (int dy = -1; dy <= 1; ++dy)
        for (int dx = -1; dx <= 1; ++dx) {
            int yy = y + dy, xx = x + dx;
            if (yy >= 0 && yy < HH && xx >= 0 && xx < WW) mx = fmaxf(mx, sig[yy*WW + xx]);
        }
    nms[i] = (s == mx) ? s : 0.f;
}

/* single block, 1024 threads: 64 iterations of argmax (tie -> lowest index) */
__global__ void k_topk(float* __restrict__ nms, float* __restrict__ junc) {
    __shared__ float sv[1024];
    __shared__ int   si[1024];
    int t = threadIdx.x;
    for (int k = 0; k < NTOP; ++k) {
        float bv = -INFINITY; int bi = 0x7FFFFFFF;
        for (int i = t; i < HWSZ; i += 1024) {
            float v = nms[i];
            if (v > bv || (v == bv && i < bi)) { bv = v; bi = i; }
        }
        sv[t] = bv; si[t] = bi; __syncthreads();
        for (int s = 512; s > 0; s >>= 1) {
            if (t < s) {
                float v2 = sv[t+s]; int i2 = si[t+s];
                if (v2 > sv[t] || (v2 == sv[t] && i2 < si[t])) { sv[t] = v2; si[t] = i2; }
            }
            __syncthreads();
        }
        if (t == 0) {
            int idx = si[0];
            junc[2*k+0] = (float)(idx / WW);
            junc[2*k+1] = (float)(idx % WW);
            nms[idx] = -INFINITY;
        }
        __syncthreads();
    }
}

__global__ void k_fill0(unsigned int* __restrict__ p, int n32) {
    int i = blockIdx.x * blockDim.x + threadIdx.x;
    if (i < n32) p[i] = 0u;
}

/* one block per line, 256 threads = 32x8 sample points; bilinear over 64 channels.
   output: xp_pad[line][spos_padded(34x10)][cin] fp16, halo pre-zeroed */
__global__ void k_sample(const float* __restrict__ feats, const float* __restrict__ junc,
                         _Float16* __restrict__ xp) {
    int line = blockIdx.x, pos = threadIdx.x;
    int l = line, i = 0, rem = NTOP - 1;
    while (l >= rem) { l -= rem; ++i; --rem; }
    int j = i + 1 + l;
    float U0 = junc[2*i], U1 = junc[2*i+1], V0 = junc[2*j], V1 = junc[2*j+1];
    float d0 = V0 - U0, d1 = V1 - U1;
    float den = fmaxf(sqrtf(d0*d0 + d1*d1), 1e-12f);
    float dir0 = d0/den, dir1 = d1/den;
    float nx0 = dir1, nx1 = -dir0;                  /* dir @ rot90 */
    int i0 = pos >> 3, i1 = pos & 7;
    float t0 = (float)i0 / 31.f;
    float t1 = (-1.f + 2.f*(float)i1/7.f) * 6.f;    /* linspace(-1,1,8)*SPAN/2 */
    float px = U0*t0 + V0*(1.f-t0) + nx0*t1;
    float py = U1*t0 + V1*(1.f-t0) + nx1*t1;
    float px0 = fminf(fmaxf(floorf(px), 0.f), WW-1.f);
    float py0 = fminf(fmaxf(floorf(py), 0.f), HH-1.f);
    float px1 = fminf(px0+1.f, WW-1.f);
    float py1 = fminf(py0+1.f, HH-1.f);
    int ix0=(int)px0, iy0=(int)py0, ix1=(int)px1, iy1=(int)py1;
    float w00=(py1-py)*(px1-px), w10=(py-py0)*(px1-px);
    float w01=(py1-py)*(px-px0), w11=(py-py0)*(px-px0);
    _Float16* dst = xp + ((size_t)line*SP1 + (i0+1)*X1P + (i1+1))*CFEAT;
    for (int c = 0; c < CFEAT; ++c) {
        const float* f = feats + (size_t)c*HWSZ;
        float v = f[iy0*WW+ix0]*w00 + f[iy1*WW+ix0]*w10 + f[iy0*WW+ix1]*w01 + f[iy1*WW+ix1]*w11;
        dst[c] = (_Float16)v;
    }
}

/* -------- weight swizzle: emit B directly in per-fragment order ----------------------
   layout: [ntile][kstep][lane(32)][e(16)]  with  k = 32*kstep + (lane>>4)*8 + (e&7) + (e>>3)*16,
           n = ntile*16 + (lane&15).  Conv K-order is tap-major: k = tap*CIN + cin.  */

__global__ void k_prep_conv_w(const float* __restrict__ w, _Float16* __restrict__ wsw,
                              int O, int CIN, int TAPS) {
    int idx = blockIdx.x * blockDim.x + threadIdx.x;
    int K = CIN * TAPS;
    if (idx >= K*O) return;
    int e = idx & 15, lane = (idx >> 4) & 31, rest = idx >> 9;
    int Ksteps = K >> 5;
    int s = rest % Ksteps, nt = rest / Ksteps;
    int k = 32*s + ((lane >> 4) << 3) + (e & 7) + ((e >> 3) << 4);
    int n = (nt << 4) + (lane & 15);
    int cin = k % CIN, tap = k / CIN;
    wsw[idx] = (_Float16)w[(n*CIN + cin)*TAPS + tap];
}

__global__ void k_prep_fc_w(const float* __restrict__ w, _Float16* __restrict__ wsw,
                            int N, int K) {
    int idx = blockIdx.x * blockDim.x + threadIdx.x;
    if (idx >= K*N) return;
    int e = idx & 15, lane = (idx >> 4) & 31, rest = idx >> 9;
    int Ksteps = K >> 5;
    int s = rest % Ksteps, nt = rest / Ksteps;
    int k = 32*s + ((lane >> 4) << 3) + (e & 7) + ((e >> 3) << 4);
    int n = (nt << 4) + (lane & 15);
    wsw[idx] = (_Float16)w[n*K + k];
}

/* ---------------- fragment helpers ---------------- */

__device__ __forceinline__ v16h load_a_frag(const _Float16* __restrict__ p) {
    v8h lo = *(const v8h*)(p);        /* k_local = kbase..kbase+7   */
    v8h hi = *(const v8h*)(p + 16);   /* k_local = 16+kbase..+23    */
    v16h a;
#pragma unroll
    for (int e = 0; e < 8; ++e) { a[e] = lo[e]; a[e+8] = hi[e]; }
    return a;
}

/* ---- conv1: implicit GEMM, M=256, N=256, K=9*64; 2x2 tile blocking per wave;
        fused BN + LReLU + 2x2 maxpool via cross-lane shuffle ---- */

__global__ void __launch_bounds__(256)
k_conv1(const _Float16* __restrict__ xp, const _Float16* __restrict__ wB,
        const float* __restrict__ b1, const float* __restrict__ g1, const float* __restrict__ be1,
        const float* __restrict__ m1, const float* __restrict__ v1, _Float16* __restrict__ p1) {
    int wv = threadIdx.x >> 5, lane = threadIdx.x & 31;
    int gw = blockIdx.x*8 + wv;                /* NLINES*8*8 waves */
    int ntp = gw & 7, mtp = (gw >> 3) & 7, line = gw >> 6;
    const _Float16* xl = xp + (size_t)line*SP1*CFEAT;
    int m = lane & 15, hi = lane >> 4, kbase = hi << 3;
    int pos = (mtp << 5) + m;                  /* ti=0 tile */
    int y = pos >> 3, x = pos & 7;
    const _Float16* bb = wB + ((size_t)(2*ntp*18)*32 + lane)*16;
    const int AOFF = 2*X1P*CFEAT;              /* +16 positions = +2 rows */
    const int BOFF = 18*512;                   /* next n-tile */
    v8f a00={}, a01={}, a10={}, a11={};
    int s = 0;
#pragma unroll 1
    for (int tap = 0; tap < 9; ++tap) {
        int spos = (y + tap/3)*X1P + (x + tap%3);
        const _Float16* ab = xl + spos*CFEAT + kbase;
#pragma unroll
        for (int c0 = 0; c0 < CFEAT; c0 += 32, ++s) {
            v16h A0 = load_a_frag(ab + c0);
            v16h A1 = load_a_frag(ab + c0 + AOFF);
            v16h B0 = *(const v16h*)(bb + (size_t)s*512);
            v16h B1 = *(const v16h*)(bb + BOFF + (size_t)s*512);
            a00 = WMMA16(A0, B0, a00);  a01 = WMMA16(A0, B1, a01);
            a10 = WMMA16(A1, B0, a10);  a11 = WMMA16(A1, B1, a11);
        }
    }
    int n = lane & 15;
#pragma unroll
    for (int tj = 0; tj < 2; ++tj) {
        int ch = ((2*ntp + tj) << 4) + n;
        float sc = rsqrtf(v1[ch] + BNEPS) * g1[ch];
        float sh = (b1[ch] - m1[ch])*sc + be1[ch];
#pragma unroll
        for (int ti = 0; ti < 2; ++ti) {
            v8f acc = ti ? (tj ? a11 : a10) : (tj ? a01 : a00);
            float v[8];
#pragma unroll
            for (int r = 0; r < 8; ++r) {
                float yv = acc[r]*sc + sh;
                v[r] = (yv >= 0.f) ? yv : NEGSL*yv;
            }
            float pm[4];
#pragma unroll
            for (int px = 0; px < 4; ++px) {
                float t = fmaxf(v[2*px], v[2*px+1]);          /* x-pair, this y-parity */
                pm[px] = fmaxf(t, __shfl_xor(t, 16, 32));     /* other y-parity lives in lane^16 */
            }
            if (hi == 0) {
                int py = (mtp << 1) + ti;                      /* pooled row = m-tile idx */
#pragma unroll
                for (int px = 0; px < 4; ++px)
                    p1[((size_t)line*SP2 + (py+1)*X2P + px + 1)*256 + ch] = (_Float16)pm[px];
            }
        }
    }
}

/* ---- conv2: M=64 (16x4), N=256, K=9*256; 2x2 blocking; fused BN+LReLU ---- */

__global__ void __launch_bounds__(256)
k_conv2(const _Float16* __restrict__ p1, const _Float16* __restrict__ wB,
        const float* __restrict__ b2, const float* __restrict__ g2, const float* __restrict__ be2,
        const float* __restrict__ m2, const float* __restrict__ v2, _Float16* __restrict__ h2) {
    int wv = threadIdx.x >> 5, lane = threadIdx.x & 31;
    int gw = blockIdx.x*8 + wv;                /* NLINES*2*8 waves */
    int ntp = gw & 7, mtp = (gw >> 3) & 1, line = gw >> 4;
    const _Float16* xl = p1 + (size_t)line*SP2*256;
    int m = lane & 15, hi = lane >> 4, kbase = hi << 3;
    int pos = (mtp << 5) + m;
    int y = pos >> 2, x = pos & 3;
    const _Float16* bb = wB + ((size_t)(2*ntp*72)*32 + lane)*16;
    const int AOFF = 4*X2P*256;                /* +16 positions = +4 rows */
    const int BOFF = 72*512;
    v8f a00={}, a01={}, a10={}, a11={};
    int s = 0;
#pragma unroll 1
    for (int tap = 0; tap < 9; ++tap) {
        int spos = (y + tap/3)*X2P + (x + tap%3);
        const _Float16* ab = xl + spos*256 + kbase;
#pragma unroll 1
        for (int c0 = 0; c0 < 256; c0 += 32, ++s) {
            v16h A0 = load_a_frag(ab + c0);
            v16h A1 = load_a_frag(ab + c0 + AOFF);
            v16h B0 = *(const v16h*)(bb + (size_t)s*512);
            v16h B1 = *(const v16h*)(bb + BOFF + (size_t)s*512);
            a00 = WMMA16(A0, B0, a00);  a01 = WMMA16(A0, B1, a01);
            a10 = WMMA16(A1, B0, a10);  a11 = WMMA16(A1, B1, a11);
        }
    }
    int n = lane & 15;
#pragma unroll
    for (int tj = 0; tj < 2; ++tj) {
        int ch = ((2*ntp + tj) << 4) + n;
        float sc = rsqrtf(v2[ch] + BNEPS) * g2[ch];
        float sh = (b2[ch] - m2[ch])*sc + be2[ch];
#pragma unroll
        for (int ti = 0; ti < 2; ++ti) {
            v8f acc = ti ? (tj ? a11 : a10) : (tj ? a01 : a00);
#pragma unroll
            for (int r = 0; r < 8; ++r) {
                float yv = acc[r]*sc + sh;
                yv = (yv >= 0.f) ? yv : NEGSL*yv;
                int opos = ((2*mtp + ti) << 4) + r + (hi << 3);
                h2[((size_t)line*64 + opos)*256 + ch] = (_Float16)yv;
            }
        }
    }
}

/* ---- conv3 (4x4 s4 VALID) + mean: 8 lines per wave (2 M-tiles), K=16*256 ---- */

__global__ void __launch_bounds__(256)
k_conv3(const _Float16* __restrict__ h2, const _Float16* __restrict__ wB,
        const float* __restrict__ b3, _Float16* __restrict__ fbuf) {
    int wv = threadIdx.x >> 5, lane = threadIdx.x & 31;
    int gw = blockIdx.x*8 + wv;                /* 252*8 waves */
    int ntp = gw & 7, mtp = gw >> 3;           /* mtp: 0..251, 8 lines each */
    int m = lane & 15, hi = lane >> 4, kbase = hi << 3;
    int lsub = m >> 2, pos = m & 3;
    const _Float16* xl = h2 + ((size_t)(mtp*8 + lsub)*64 + pos*16)*256 + kbase;
    const _Float16* bb = wB + ((size_t)(2*ntp*128)*32 + lane)*16;
    const int AOFF = 4*64*256;                 /* +4 lines */
    const int BOFF = 128*512;
    v8f a00={}, a01={}, a10={}, a11={};
    int s = 0;
#pragma unroll 1
    for (int t = 0; t < 16; ++t) {
        const _Float16* ab = xl + t*256;
#pragma unroll 1
        for (int c0 = 0; c0 < 256; c0 += 32, ++s) {
            v16h A0 = load_a_frag(ab + c0);
            v16h A1 = load_a_frag(ab + c0 + AOFF);
            v16h B0 = *(const v16h*)(bb + (size_t)s*512);
            v16h B1 = *(const v16h*)(bb + BOFF + (size_t)s*512);
            a00 = WMMA16(A0, B0, a00);  a01 = WMMA16(A0, B1, a01);
            a10 = WMMA16(A1, B0, a10);  a11 = WMMA16(A1, B1, a11);
        }
    }
    /* mean over 4 positions: rows 4*ml..4*ml+3; lo lanes own ml 0,1; hi lanes ml 2,3 */
    int n = lane & 15, mlb = hi << 1;
#pragma unroll
    for (int tj = 0; tj < 2; ++tj) {
        int ch = ((2*ntp + tj) << 4) + n;
        float bb3 = b3[ch];
#pragma unroll
        for (int ti = 0; ti < 2; ++ti) {
            v8f acc = ti ? (tj ? a11 : a10) : (tj ? a01 : a00);
            float sA = acc[0] + acc[1] + acc[2] + acc[3];
            float sB = acc[4] + acc[5] + acc[6] + acc[7];
            size_t line0 = (size_t)(2*mtp + ti)*4;
            fbuf[(line0 + mlb + 0)*256 + ch] = (_Float16)(0.25f*sA + bb3);
            fbuf[(line0 + mlb + 1)*256 + ch] = (_Float16)(0.25f*sB + bb3);
        }
    }
}

/* ---------------- generic fp16 GEMM + bias + ReLU, 2x2 blocking (MLP hidden) -------- */

__global__ void __launch_bounds__(256)
k_gemm_relu(const _Float16* __restrict__ A, const _Float16* __restrict__ B,
            const float* __restrict__ bias, _Float16* __restrict__ C, int K, int N) {
    int wv = threadIdx.x >> 5, lane = threadIdx.x & 31;
    int gw = blockIdx.x*8 + wv;
    int ntp2 = N >> 5, Ksteps = K >> 5;
    int ntp = gw % ntp2, mtp = gw / ntp2;
    int m = lane & 15, hi = lane >> 4, kbase = hi << 3;
    const _Float16* Arow = A + (size_t)((mtp << 5) + m)*K + kbase;
    const _Float16* bb   = B + ((size_t)(2*ntp*Ksteps)*32 + lane)*16;
    const int AOFF = 16*K;                     /* +16 rows */
    const int BOFF = Ksteps*512;
    v8f a00={}, a01={}, a10={}, a11={};
#pragma unroll 1
    for (int s = 0; s < Ksteps; ++s) {
        v16h A0 = load_a_frag(Arow + s*32);
        v16h A1 = load_a_frag(Arow + s*32 + AOFF);
        v16h B0 = *(const v16h*)(bb + (size_t)s*512);
        v16h B1 = *(const v16h*)(bb + BOFF + (size_t)s*512);
        a00 = WMMA16(A0, B0, a00);  a01 = WMMA16(A0, B1, a01);
        a10 = WMMA16(A1, B0, a10);  a11 = WMMA16(A1, B1, a11);
    }
    int n = lane & 15;
#pragma unroll
    for (int tj = 0; tj < 2; ++tj) {
        int col = ((2*ntp + tj) << 4) + n;
        float bbv = bias[col];
#pragma unroll
        for (int ti = 0; ti < 2; ++ti) {
            v8f acc = ti ? (tj ? a11 : a10) : (tj ? a01 : a00);
#pragma unroll
            for (int r = 0; r < 8; ++r) {
                float yv = acc[r] + bbv;
                yv = (yv > 0.f) ? yv : 0.f;
                int row = ((2*mtp + ti) << 4) + r + (hi << 3);
                C[(size_t)row*N + col] = (_Float16)yv;
            }
        }
    }
}

/* ---------------- final heads: N=7 logits + N=1 width (tiny -> VALU dots) ------------ */

__global__ void k_heads(const _Float16* __restrict__ x2c, const _Float16* __restrict__ x2w,
                        const float* __restrict__ wc3, const float* __restrict__ bc3,
                        const float* __restrict__ ww3, const float* __restrict__ bw3,
                        float* __restrict__ out) {
    int t = blockIdx.x * blockDim.x + threadIdx.x;
    if (t >= NLINES*8) return;
    int line = t >> 3, h = t & 7;
    if (h < 7) {
        const _Float16* xr = x2c + (size_t)line*512;
        const float*    wr = wc3 + (size_t)h*512;
        float s = 0.f;
        for (int k = 0; k < 512; ++k) s += (float)xr[k] * wr[k];
        out[line*7 + h] = s + bc3[h];
    } else {
        const _Float16* xr = x2w + (size_t)line*512;
        float s = 0.f;
        for (int k = 0; k < 512; ++k) s += (float)xr[k] * ww3[k];
        out[NLINES*7 + line] = s + bw3[0];
    }
}

/* ---------------- host-side launch ---------------- */

extern "C" void kernel_launch(void* const* d_in, const int* in_sizes, int n_in,
                              void* d_out, int out_size, void* d_ws, size_t ws_size,
                              hipStream_t stream) {
    const float* feats = (const float*)d_in[0];
    const float* jloc  = (const float*)d_in[1];
    const float* w1  = (const float*)d_in[2];  const float* b1  = (const float*)d_in[3];
    const float* g1  = (const float*)d_in[4];  const float* be1 = (const float*)d_in[5];
    const float* m1  = (const float*)d_in[6];  const float* v1  = (const float*)d_in[7];
    const float* w2  = (const float*)d_in[8];  const float* b2  = (const float*)d_in[9];
    const float* g2  = (const float*)d_in[10]; const float* be2 = (const float*)d_in[11];
    const float* m2  = (const float*)d_in[12]; const float* v2  = (const float*)d_in[13];
    const float* w3  = (const float*)d_in[14]; const float* b3  = (const float*)d_in[15];
    const float* wc1 = (const float*)d_in[16]; const float* bc1 = (const float*)d_in[17];
    const float* wc2 = (const float*)d_in[18]; const float* bc2 = (const float*)d_in[19];
    const float* wc3 = (const float*)d_in[20]; const float* bc3 = (const float*)d_in[21];
    const float* ww1 = (const float*)d_in[22]; const float* bw1 = (const float*)d_in[23];
    const float* ww2 = (const float*)d_in[24]; const float* bw2 = (const float*)d_in[25];
    const float* ww3 = (const float*)d_in[26]; const float* bw3 = (const float*)d_in[27];
    float* out = (float*)d_out;

    char* ws = (char*)d_ws;
    size_t off = 0;
    auto take = [&](size_t bytes) -> char* {
        char* p = ws + off;
        off = (off + bytes + 255) & ~(size_t)255;
        return p;
    };

    const size_t XP_HALVES = (size_t)NLINES*SP1*CFEAT;   /* padded */
    const size_t P1_HALVES = (size_t)NLINES*SP2*256;     /* padded */

    float*    sig  = (float*)take((size_t)HWSZ*4);
    float*    nms  = (float*)take((size_t)HWSZ*4);
    float*    junc = (float*)take((size_t)NTOP*2*4);
    _Float16* xp   = (_Float16*)take(XP_HALVES*2);       /* ~88 MB */
    _Float16* p1   = (_Float16*)take(P1_HALVES*2);       /* ~111 MB */
    _Float16* h2   = xp;                 /* xp dead after conv1; 33M halves fit */
    _Float16* fbuf = (_Float16*)take((size_t)NLINES*256*2);
    _Float16* x1c  = (_Float16*)take((size_t)NLINES*512*2);
    _Float16* x2c  = (_Float16*)take((size_t)NLINES*512*2);
    _Float16* x1w  = (_Float16*)take((size_t)NLINES*512*2);
    _Float16* x2w  = (_Float16*)take((size_t)NLINES*512*2);
    _Float16* wB1  = (_Float16*)take((size_t)576*256*2);
    _Float16* wB2  = (_Float16*)take((size_t)2304*256*2);
    _Float16* wB3  = (_Float16*)take((size_t)4096*256*2);
    _Float16* wBc1 = (_Float16*)take((size_t)256*512*2);
    _Float16* wBc2 = (_Float16*)take((size_t)512*512*2);
    _Float16* wBw1 = (_Float16*)take((size_t)256*512*2);
    _Float16* wBw2 = (_Float16*)take((size_t)512*512*2);
    (void)in_sizes; (void)n_in; (void)out_size; (void)ws_size;

    /* zero padded activation buffers (halos must be 0 every call) */
    {
        int n32a = (int)(XP_HALVES/2), n32b = (int)(P1_HALVES/2);
        k_fill0<<<(n32a + 255)/256, 256, 0, stream>>>((unsigned int*)xp, n32a);
        k_fill0<<<(n32b + 255)/256, 256, 0, stream>>>((unsigned int*)p1, n32b);
    }

    /* weight swizzle (fragment order) */
    k_prep_conv_w<<<(576*256 + 255)/256,  256, 0, stream>>>(w1, wB1, 256, 64, 9);
    k_prep_conv_w<<<(2304*256 + 255)/256, 256, 0, stream>>>(w2, wB2, 256, 256, 9);
    k_prep_conv_w<<<(4096*256 + 255)/256, 256, 0, stream>>>(w3, wB3, 256, 256, 16);
    k_prep_fc_w<<<(256*512 + 255)/256, 256, 0, stream>>>(wc1, wBc1, 512, 256);
    k_prep_fc_w<<<(512*512 + 255)/256, 256, 0, stream>>>(wc2, wBc2, 512, 512);
    k_prep_fc_w<<<(256*512 + 255)/256, 256, 0, stream>>>(ww1, wBw1, 512, 256);
    k_prep_fc_w<<<(512*512 + 255)/256, 256, 0, stream>>>(ww2, wBw2, 512, 512);

    /* detection + sampling */
    k_sigmoid<<<HWSZ/256, 256, 0, stream>>>(jloc, sig);
    k_nms<<<HWSZ/256, 256, 0, stream>>>(sig, nms);
    k_topk<<<1, 1024, 0, stream>>>(nms, junc);
    k_sample<<<NLINES, 256, 0, stream>>>(feats, junc, xp);

    /* conv head: WMMA implicit GEMMs, 2x2 register blocking */
    k_conv1<<<(NLINES*8*8)/8, 256, 0, stream>>>(xp, wB1, b1, g1, be1, m1, v1, p1);
    k_conv2<<<(NLINES*2*8)/8, 256, 0, stream>>>(p1, wB2, b2, g2, be2, m2, v2, h2);
    k_conv3<<<(252*8)/8, 256, 0, stream>>>(h2, wB3, b3, fbuf);

    /* MLPs: M=2016 (=63 tile-pairs), N=512 (=16 tile-pairs) */
    k_gemm_relu<<<(63*16)/8, 256, 0, stream>>>(fbuf, wBc1, bc1, x1c, 256, 512);
    k_gemm_relu<<<(63*16)/8, 256, 0, stream>>>(x1c,  wBc2, bc2, x2c, 512, 512);
    k_gemm_relu<<<(63*16)/8, 256, 0, stream>>>(fbuf, wBw1, bw1, x1w, 256, 512);
    k_gemm_relu<<<(63*16)/8, 256, 0, stream>>>(x1w,  wBw2, bw2, x2w, 512, 512);

    k_heads<<<(NLINES*8 + 255)/256, 256, 0, stream>>>(x2c, x2w, wc3, bc3, ww3, bw3, out);
}
